// Encoder_19963007992483
// MI455X (gfx1250) — matrix operand order repaired and checked by reading, not compiled, not run on hardware
//
#include <hip/hip_runtime.h>
#include <cstddef>
#include <cstdint>

// ---------------------------------------------------------------------------
// Encoder: 2x (ConvAttention + ChannelMLP), B=16, D=T=1024, KW=31.
// Conv implicit-GEMM + channel-MLP GEMM on v_wmma_f32_16x16x32_bf16.
// Activation tiles staged into LDS via TDM (tensor_load_to_lds) when the tile
// is halo-free, else per-lane global_load_async_to_lds_b128 (both guarded with
// __has_builtin; falls back to VGPR staging if the toolchain lacks them).
// ---------------------------------------------------------------------------

constexpr int NB  = 16;    // batch
constexpr int ND  = 1024;  // channels == time
constexpr int NKW = 31;    // conv taps
constexpr int HALO = 15;   // (NKW-1)/2
constexpr float LN_EPS = 1e-3f;

constexpr int TM = 128;    // block tile M
constexpr int TN = 128;    // block tile N
constexpr int KC = 64;     // K chunk staged in LDS (two wmma K-steps of 32)

typedef __attribute__((ext_vector_type(16))) __bf16 v16bf;
typedef __attribute__((ext_vector_type(8)))  float  v8f;

union Frag {
    uint4 q[2];
    v16bf v;
};

// ---- gfx1250 async/TDM feature detection (always-compiling guards) --------
#if defined(__AMDGCN__) && __has_builtin(__builtin_amdgcn_global_load_async_to_lds_b128)
#define HAVE_ASYNC_B128 1
#else
#define HAVE_ASYNC_B128 0
#endif

#if defined(__AMDGCN__) && __has_builtin(__builtin_amdgcn_tensor_load_to_lds)
#define HAVE_TDM 1
#else
#define HAVE_TDM 0
#endif

#if HAVE_TDM
typedef unsigned int uint32x4 __attribute__((ext_vector_type(4)));
typedef int          int32x4  __attribute__((ext_vector_type(4)));
typedef int          int32x8  __attribute__((ext_vector_type(8)));
#endif

// GCC-style int4 vector: matches the async builtin's pointee type exactly.
typedef int v4i_vs __attribute__((vector_size(16)));

// 16-byte global -> LDS copy; async (ASYNCcnt) when available.
__device__ __forceinline__ void g2l_b128(const void* gsrc, void* ldst) {
#if HAVE_ASYNC_B128
    __builtin_amdgcn_global_load_async_to_lds_b128(
        (__attribute__((address_space(1))) v4i_vs*)(unsigned long long)(uintptr_t)gsrc,
        (__attribute__((address_space(3))) v4i_vs*)(unsigned int)(uintptr_t)ldst,
        0, 0);
#else
    *(uint4*)ldst = *(const uint4*)gsrc;
#endif
}

__device__ __forceinline__ void wait_async0() {
#if HAVE_ASYNC_B128
# if __has_builtin(__builtin_amdgcn_s_wait_asynccnt)
    __builtin_amdgcn_s_wait_asynccnt(0);
# else
    asm volatile("s_wait_asynccnt 0" ::: "memory");
# endif
#endif
}

__device__ __forceinline__ void wait_tensor0() {
#if HAVE_TDM
# if __has_builtin(__builtin_amdgcn_s_wait_tensorcnt)
    __builtin_amdgcn_s_wait_tensorcnt(0);
# else
    asm volatile("s_wait_tensorcnt 0" ::: "memory");
# endif
#endif
}

#if HAVE_TDM
// Strided-2D bf16 tile copy global->LDS through the Tensor Data Mover.
// D# packed per CDNA5 ISA ch.8: group0 {count, lds_addr, global_addr, type=2},
// group1 {data_size=2B, tensor_dim0/1, tile_dim0/1, dim0_stride}. Groups 2/3
// zero (<=2D tensor). Tracked by TENSORcnt.
__device__ __forceinline__ void tdm_load_2d_bf16(const void* gsrc, void* ldst,
                                                 unsigned tile_w_elems,
                                                 unsigned tile_h_rows,
                                                 unsigned row_stride_elems) {
    const unsigned long long ga = (unsigned long long)(uintptr_t)gsrc;
    const unsigned la = (unsigned)(uintptr_t)ldst;   // flat[31:0] == LDS offset
    uint32x4 g0;
    g0[0] = 1u;                                         // count=1, user D#
    g0[1] = la;                                         // lds_addr
    g0[2] = (unsigned)(ga & 0xFFFFFFFFull);             // global_addr[31:0]
    g0[3] = (unsigned)((ga >> 32) & 0x01FFFFFFull)      // global_addr[56:32]
          | (2u << 30);                                 // type=2 ("image")
    int32x8 g1;
    g1[0] = (int)(1u << 16);                            // data_size=1 -> 2 bytes
    g1[1] = (int)((tile_w_elems & 0xFFFFu) << 16);      // tensor_dim0[15:0]
    g1[2] = (int)((tile_w_elems >> 16) | ((tile_h_rows & 0xFFFFu) << 16));
    g1[3] = (int)((tile_h_rows >> 16) | ((tile_w_elems & 0xFFFFu) << 16)); // tile_dim0
    g1[4] = (int)(tile_h_rows & 0xFFFFu);               // tile_dim1 (tile_dim2=0)
    g1[5] = (int)row_stride_elems;                      // tensor_dim0_stride[31:0]
    g1[6] = 0;
    g1[7] = 0;
    int32x4 gz = {0, 0, 0, 0};
#if __clang_major__ >= 23
    int32x8 gz8 = {0, 0, 0, 0, 0, 0, 0, 0};
    __builtin_amdgcn_tensor_load_to_lds(g0, g1, gz, gz, gz8, 0);
#else
    __builtin_amdgcn_tensor_load_to_lds(g0, g1, gz, gz, 0);
#endif
}
#endif  // HAVE_TDM

__device__ __forceinline__ unsigned short f2bf(float f) {
    unsigned int u = __float_as_uint(f);
    u += 0x7FFFu + ((u >> 16) & 1u);
    return (unsigned short)(u >> 16);
}

__device__ __forceinline__ float gelu_exact(float y) {
    return 0.5f * y * (1.0f + erff(y * 0.70710678118654752f));
}

// ---------------------------------------------------------------------------
// LayerNorm over last axis (D=1024) + cast to bf16.  One block per row.
// ---------------------------------------------------------------------------
__global__ __launch_bounds__(256)
void ln_cast_kernel(const float* __restrict__ x,
                    const float* __restrict__ gamma,
                    const float* __restrict__ beta,
                    unsigned short* __restrict__ xn) {
    const int row = blockIdx.x;                 // 0 .. NB*ND-1
    const float* xr = x + (size_t)row * ND;
    __shared__ float red[256];
    const int tid = threadIdx.x;

    float s = 0.f;
    for (int i = tid; i < ND; i += 256) s += xr[i];
    red[tid] = s; __syncthreads();
    for (int off = 128; off > 0; off >>= 1) {
        if (tid < off) red[tid] += red[tid + off];
        __syncthreads();
    }
    const float mu = red[0] * (1.0f / ND);
    __syncthreads();

    float v = 0.f;
    for (int i = tid; i < ND; i += 256) { float d = xr[i] - mu; v += d * d; }
    red[tid] = v; __syncthreads();
    for (int off = 128; off > 0; off >>= 1) {
        if (tid < off) red[tid] += red[tid + off];
        __syncthreads();
    }
    const float rs = rsqrtf(red[0] * (1.0f / ND) + LN_EPS);

    unsigned short* xo = xn + (size_t)row * ND;
    for (int i = tid; i < ND; i += 256) {
        float y = (xr[i] - mu) * rs * gamma[i] + beta[i];
        xo[i] = f2bf(y);
    }
}

// ---------------------------------------------------------------------------
// Conv1D 'same' (KW=31) as implicit GEMM:
//   out[b,t,f] = gelu( sum_{w,i} xn[b, t+w-15, i] * K[w,i,f] + bias[f] ) + xres
// One block = 128x128 (t,f) tile per batch; 8 waves = 4(M) x 2(N); each wave
// owns 2x4 16x16 fp32 accumulator tiles.
// ---------------------------------------------------------------------------
__global__ __launch_bounds__(256)
void conv_gemm_kernel(const unsigned short* __restrict__ xn,  // [NB, ND, ND] bf16
                      const float* __restrict__ Kw,           // [NKW, ND, ND]
                      const float* __restrict__ bias,         // [ND]
                      const float* __restrict__ xres,         // [NB, ND, ND]
                      float* __restrict__ out) {              // [NB, ND, ND]
    __shared__ alignas(16) unsigned short As[TM][KC];  // [m][k] row-major
    __shared__ alignas(16) unsigned short Bs[TN][KC];  // [n][k] (transposed)

    const int blk   = blockIdx.x;
    const int batch = blk >> 6;
    const int rem   = blk & 63;
    const int tT    = (rem >> 3) * TM;   // time-tile origin
    const int tF    = (rem & 7)  * TN;   // feature-tile origin

    const unsigned short* xnB = xn + (size_t)batch * ND * ND;
    const int tid  = threadIdx.x;
    const int lane = tid & 31;
    const int wv   = tid >> 5;
    const int wr   = wv & 3;             // 0..3 -> M offset wr*32
    const int wc   = wv >> 2;            // 0..1 -> N offset wc*64
    const int half = lane >> 4;
    const int l15  = lane & 15;

    v8f acc[2][4];
    for (int mt = 0; mt < 2; ++mt)
        for (int nt = 0; nt < 4; ++nt)
            acc[mt][nt] = (v8f)0.0f;

    constexpr int NKC = NKW * (ND / KC);  // 496 K-chunks
    for (int kc = 0; kc < NKC; ++kc) {
        const int w      = kc >> 4;          // 16 chunks per tap
        const int i0     = (kc & 15) * KC;
        const int tshift = w - HALO;
        const int tBase  = tT + tshift;
        const bool interior = (tBase >= 0) && (tBase + TM <= ND);

        // ---- stage A: As[m][k] = xn[b, tT+m+tshift, i0+k] (zero-padded) ----
#if HAVE_TDM
        if (interior) {
            // whole 128x64 bf16 tile in one TDM strided-2D descriptor
            if (wv == 0)
                tdm_load_2d_bf16(xnB + (size_t)tBase * ND + i0, &As[0][0],
                                 KC, TM, ND);
        } else
#endif
        {
            #pragma unroll
            for (int j = 0; j < 4; ++j) {
                int c  = tid + 256 * j;          // 1024 16-byte chunks
                int m  = c >> 3;
                int k8 = (c & 7) * 8;
                int t  = tBase + m;
                void* ldst = (void*)&As[m][k8];
                if (t >= 0 && t < ND) {
                    g2l_b128(xnB + (size_t)t * ND + i0 + k8, ldst);
                } else {
                    *(uint4*)ldst = make_uint4(0u, 0u, 0u, 0u);
                }
            }
        }
        // ---- stage B (transpose + fp32->bf16): Bs[n][k] = K[w, i0+k, tF+n] ----
        const float* Kbase = Kw + ((size_t)w * ND + i0) * ND + tF;
        #pragma unroll
        for (int j = 0; j < 8; ++j) {
            int g  = tid + 256 * j;              // 2048 float4 groups
            int k  = g >> 5;
            int n4 = (g & 31) * 4;
            const float4 f4 = *(const float4*)(Kbase + (size_t)k * ND + n4);
            Bs[n4 + 0][k] = f2bf(f4.x);
            Bs[n4 + 1][k] = f2bf(f4.y);
            Bs[n4 + 2][k] = f2bf(f4.z);
            Bs[n4 + 3][k] = f2bf(f4.w);
        }
        // prefetch next weight chunk toward L2 (global_prefetch_b8)
        if (kc + 1 < NKC) {
            int w2 = (kc + 1) >> 4, i02 = ((kc + 1) & 15) * KC;
            __builtin_prefetch(Kw + ((size_t)w2 * ND + i02) * ND + tF + tid * 8, 0, 1);
        }
        wait_tensor0();
        wait_async0();
        __syncthreads();

        // ---- two wmma K-steps of 32 ----
        #pragma unroll
        for (int ks = 0; ks < KC; ks += 32) {
            Frag af[2], bf[4];
            const int akb = ks + (half ? 8 : 0);   // A frag: {0-7,16-23} | {8-15,24-31}
            #pragma unroll
            for (int mt = 0; mt < 2; ++mt) {
                int m = wr * 32 + mt * 16 + l15;
                af[mt].q[0] = *(const uint4*)(&As[m][akb]);
                af[mt].q[1] = *(const uint4*)(&As[m][akb + 16]);
            }
            const int bkb = ks + (half ? 8 : 0);   // B frag: {0-7,16-23} | {8-15,24-31}
            #pragma unroll
            for (int nt = 0; nt < 4; ++nt) {
                int n = wc * 64 + nt * 16 + l15;
                bf[nt].q[0] = *(const uint4*)(&Bs[n][bkb]);
                bf[nt].q[1] = *(const uint4*)(&Bs[n][bkb + 16]);
            }
            #pragma unroll
            for (int mt = 0; mt < 2; ++mt)
                #pragma unroll
                for (int nt = 0; nt < 4; ++nt)
                    acc[mt][nt] = __builtin_amdgcn_wmma_f32_16x16x32_bf16(
                        false, af[mt].v, false, bf[nt].v,
                        (short)0, acc[mt][nt], false, false);
        }
        __syncthreads();
    }

    // ---- epilogue: bias + exact GELU + residual ----
    #pragma unroll
    for (int mt = 0; mt < 2; ++mt) {
        #pragma unroll
        for (int nt = 0; nt < 4; ++nt) {
            const int f = tF + wc * 64 + nt * 16 + l15;
            const float bf_ = bias[f];
            #pragma unroll
            for (int j = 0; j < 8; ++j) {
                int t = tT + wr * 32 + mt * 16 + j + half * 8;
                size_t idx = ((size_t)batch * ND + t) * ND + f;
                float y = acc[mt][nt][j] + bf_;
                out[idx] = gelu_exact(y) + xres[idx];
            }
        }
    }
}

// ---------------------------------------------------------------------------
// ChannelMLP: out[b,i,f] = gelu( sum_t xn[b,t,i] * W[t,f] + wb[f] ) + xres[b,i,f]
// GEMM with A[m=i][k=t] = xn[t,i] (transposed access), B[k=t][n=f] = W[t,f].
// ---------------------------------------------------------------------------
__global__ __launch_bounds__(256)
void mlp_gemm_kernel(const unsigned short* __restrict__ xn,   // [NB, ND, ND] bf16
                     const float* __restrict__ W,             // [ND, ND]
                     const float* __restrict__ wb,            // [ND]
                     const float* __restrict__ xres,          // [NB, ND, ND]
                     float* __restrict__ out) {               // [NB, ND, ND]
    __shared__ alignas(16) unsigned short As[TM][KC];
    __shared__ alignas(16) unsigned short Bs[TN][KC];

    const int blk   = blockIdx.x;
    const int batch = blk >> 6;
    const int rem   = blk & 63;
    const int tI    = (rem >> 3) * TM;   // channel-i tile (GEMM M)
    const int tF    = (rem & 7)  * TN;   // feature tile   (GEMM N)

    const unsigned short* xnB = xn + (size_t)batch * ND * ND;
    const int tid  = threadIdx.x;
    const int lane = tid & 31;
    const int wv   = tid >> 5;
    const int wr   = wv & 3;
    const int wc   = wv >> 2;
    const int half = lane >> 4;
    const int l15  = lane & 15;

    v8f acc[2][4];
    for (int mt = 0; mt < 2; ++mt)
        for (int nt = 0; nt < 4; ++nt)
            acc[mt][nt] = (v8f)0.0f;

    constexpr int NKC = ND / KC;  // 16 chunks
    for (int kc = 0; kc < NKC; ++kc) {
        const int k0 = kc * KC;

        // ---- stage A transposed: As[m][k] = xn[b, k0+k, tI+m] ----
        #pragma unroll
        for (int j = 0; j < 4; ++j) {
            int g  = tid + 256 * j;          // 1024 groups of 8 elems along m
            int k  = g >> 4;
            int m8 = (g & 15) * 8;
            const unsigned short* src = xnB + (size_t)(k0 + k) * ND + tI + m8;
            uint4 v = *(const uint4*)src;
            unsigned short tmp[8];
            *(uint4*)tmp = v;
            #pragma unroll
            for (int e = 0; e < 8; ++e) As[m8 + e][k] = tmp[e];
        }
        // ---- stage B (transpose + cast): Bs[n][k] = W[k0+k, tF+n] ----
        const float* Wbase = W + (size_t)k0 * ND + tF;
        #pragma unroll
        for (int j = 0; j < 8; ++j) {
            int g  = tid + 256 * j;
            int k  = g >> 5;
            int n4 = (g & 31) * 4;
            const float4 f4 = *(const float4*)(Wbase + (size_t)k * ND + n4);
            Bs[n4 + 0][k] = f2bf(f4.x);
            Bs[n4 + 1][k] = f2bf(f4.y);
            Bs[n4 + 2][k] = f2bf(f4.z);
            Bs[n4 + 3][k] = f2bf(f4.w);
        }
        if (kc + 1 < NKC)
            __builtin_prefetch(W + (size_t)(k0 + KC) * ND + tF + tid * 8, 0, 1);
        __syncthreads();

        #pragma unroll
        for (int ks = 0; ks < KC; ks += 32) {
            Frag af[2], bf[4];
            const int akb = ks + (half ? 8 : 0);
            #pragma unroll
            for (int mt = 0; mt < 2; ++mt) {
                int m = wr * 32 + mt * 16 + l15;
                af[mt].q[0] = *(const uint4*)(&As[m][akb]);
                af[mt].q[1] = *(const uint4*)(&As[m][akb + 16]);
            }
            const int bkb = ks + (half ? 8 : 0);
            #pragma unroll
            for (int nt = 0; nt < 4; ++nt) {
                int n = wc * 64 + nt * 16 + l15;
                bf[nt].q[0] = *(const uint4*)(&Bs[n][bkb]);
                bf[nt].q[1] = *(const uint4*)(&Bs[n][bkb + 16]);
            }
            #pragma unroll
            for (int mt = 0; mt < 2; ++mt)
                #pragma unroll
                for (int nt = 0; nt < 4; ++nt)
                    acc[mt][nt] = __builtin_amdgcn_wmma_f32_16x16x32_bf16(
                        false, af[mt].v, false, bf[nt].v,
                        (short)0, acc[mt][nt], false, false);
        }
        __syncthreads();
    }

    #pragma unroll
    for (int mt = 0; mt < 2; ++mt) {
        #pragma unroll
        for (int nt = 0; nt < 4; ++nt) {
            const int f = tF + wc * 64 + nt * 16 + l15;
            const float bf_ = wb[f];
            #pragma unroll
            for (int j = 0; j < 8; ++j) {
                int i = tI + wr * 32 + mt * 16 + j + half * 8;
                size_t idx = ((size_t)batch * ND + i) * ND + f;
                float y = acc[mt][nt][j] + bf_;
                out[idx] = gelu_exact(y) + xres[idx];
            }
        }
    }
}

// ---------------------------------------------------------------------------
extern "C" void kernel_launch(void* const* d_in, const int* in_sizes, int n_in,
                              void* d_out, int out_size, void* d_ws, size_t ws_size,
                              hipStream_t stream) {
    (void)in_sizes; (void)n_in; (void)out_size; (void)ws_size;

    const float* x       = (const float*)d_in[0];
    const float* ln_g1   = (const float*)d_in[1];
    const float* ln_b1   = (const float*)d_in[2];
    const float* conv_k1 = (const float*)d_in[3];
    const float* conv_b1 = (const float*)d_in[4];
    const float* ln_gm1  = (const float*)d_in[5];
    const float* ln_bm1  = (const float*)d_in[6];
    const float* W1      = (const float*)d_in[7];
    const float* Wb1     = (const float*)d_in[8];
    const float* ln_g2   = (const float*)d_in[9];
    const float* ln_b2   = (const float*)d_in[10];
    const float* conv_k2 = (const float*)d_in[11];
    const float* conv_b2 = (const float*)d_in[12];
    const float* ln_gm2  = (const float*)d_in[13];
    const float* ln_bm2  = (const float*)d_in[14];
    const float* W2      = (const float*)d_in[15];
    const float* Wb2     = (const float*)d_in[16];
    float* out = (float*)d_out;

    // workspace: [ xn bf16 : NB*ND*ND*2 = 32MB ][ xA fp32 : NB*ND*ND*4 = 64MB ]
    unsigned short* xn = (unsigned short*)d_ws;
    float* xA = (float*)((char*)d_ws + (size_t)NB * ND * ND * sizeof(unsigned short));

    const dim3 lnGrid(NB * ND), blk256(256);
    const dim3 gemmGrid(NB * (ND / TM) * (ND / TN));   // 16 * 8 * 8 = 1024 blocks

    // Layer 1: ConvAttention  (x -> xA)
    ln_cast_kernel<<<lnGrid, blk256, 0, stream>>>(x, ln_g1, ln_b1, xn);
    conv_gemm_kernel<<<gemmGrid, blk256, 0, stream>>>(xn, conv_k1, conv_b1, x, xA);
    // Layer 1: ChannelMLP     (xA -> out)
    ln_cast_kernel<<<lnGrid, blk256, 0, stream>>>(xA, ln_gm1, ln_bm1, xn);
    mlp_gemm_kernel<<<gemmGrid, blk256, 0, stream>>>(xn, W1, Wb1, xA, out);
    // Layer 2: ConvAttention  (out -> xA)
    ln_cast_kernel<<<lnGrid, blk256, 0, stream>>>(out, ln_g2, ln_b2, xn);
    conv_gemm_kernel<<<gemmGrid, blk256, 0, stream>>>(xn, conv_k2, conv_b2, out, xA);
    // Layer 2: ChannelMLP     (xA -> out)
    ln_cast_kernel<<<lnGrid, blk256, 0, stream>>>(xA, ln_gm2, ln_bm2, xn);
    mlp_gemm_kernel<<<gemmGrid, blk256, 0, stream>>>(xn, W2, Wb2, xA, out);
}
